// InterModalLoss_29858612642238
// MI455X (gfx1250) — matrix operand order repaired and checked by reading, not compiled, not run on hardware
//
#include <hip/hip_runtime.h>
#include <hip/hip_bf16.h>

typedef __attribute__((ext_vector_type(16))) __bf16 v16bf;
typedef __attribute__((ext_vector_type(8)))  float  v8f;
typedef __attribute__((ext_vector_type(4)))  unsigned int v4u;
typedef __attribute__((ext_vector_type(4)))  int    v4i;
typedef __attribute__((ext_vector_type(8)))  int    v8i;

#define B_SZ 4096
#define D_SZ 1024
#define TEMP 0.2f
#define HGAMMA 0.5f

// GEMM tiling
#define KC    32     // K elements staged per buffer
#define PITCH 40     // 32 bf16 + 4-DWORD TDM pad -> 80B rows, conflict-free b128 LDS reads

union FragAB { v16bf v; uint4 q[2]; };

__device__ __forceinline__ float softplus_f(float y) {
    return fmaxf(y, 0.0f) + __logf(1.0f + __expf(-fabsf(y)));
}

// ---- Tensor Data Mover: DMA one 128x32 bf16 tile (row-major, K fast) into LDS ----
// D# per CDNA5 ISA ch.8: group0 {count, lds_addr, global_addr(57b), type=2},
// group1 {data_size=2B, pad_enable, pad_interval=16DW(64B row), pad_amount=4DW,
//         tensor_dim0=1024, tensor_dim1=4096, tile_dim0=32, tile_dim1=128,
//         tensor_dim0_stride=1024}, remaining groups zero (2-D tile).
// This toolchain's builtin is the 6-arg clang-23 form:
//   (uint32x4, int32x8, int32x4, int32x4, int32x8, i32 cpol)
__device__ __forceinline__ void tdm_load_tile(const unsigned short* gptr, unsigned lds_off) {
    unsigned long long ga = (unsigned long long)(uintptr_t)gptr;
    v4u g0;
    g0.x = 1u;                                               // count=1, user descriptor
    g0.y = lds_off;                                          // LDS byte address
    g0.z = (unsigned)ga;                                     // global_addr[31:0]
    g0.w = (unsigned)((ga >> 32) & 0x1FFFFFFu) | 0x80000000u; // addr[56:32] | type=2
    v8i g1;
    g1[0] = (int)((1u << 16)      // data_size = 2 bytes
                | (1u << 20)      // pad_enable
                | (3u << 22)      // pad_interval code 3 -> every 16 DWORDs (64B row)
                | (3u << 25));    // pad_amount code 3 -> 4 DWORDs (16B)
    g1[1] = (int)((unsigned)D_SZ << 16);          // tensor_dim0[15:0] in bits 63:48
    g1[2] = (int)((unsigned)B_SZ << 16);          // tensor_dim1[15:0] in bits 95:80
    g1[3] = (int)((unsigned)KC   << 16);          // tile_dim0 in bits 127:112
    g1[4] = 128;                                  // tile_dim1 in bits 143:128
    g1[5] = D_SZ;                                 // tensor_dim0_stride[31:0]
    g1[6] = 0;
    g1[7] = 0;
    v4i gz4 = {0, 0, 0, 0};
    v8i gz8 = {0, 0, 0, 0, 0, 0, 0, 0};
    __builtin_amdgcn_tensor_load_to_lds(g0, g1, gz4, gz4, gz8, 0);
}

// ---------------- Kernel 1: row L2-normalize f32 -> bf16 ----------------
__global__ __launch_bounds__(256) void norm_kernel(const float* __restrict__ src,
                                                   unsigned short* __restrict__ dst) {
    const int row = blockIdx.x;
    const int t = threadIdx.x;
    const float* r = src + (size_t)row * D_SZ;
    float4 v = *(const float4*)(r + t * 4);
    float ss = v.x * v.x + v.y * v.y + v.z * v.z + v.w * v.w;

    __shared__ float red[256];
    red[t] = ss;
    __syncthreads();
#pragma unroll
    for (int s = 128; s > 0; s >>= 1) {
        if (t < s) red[t] += red[t + s];
        __syncthreads();
    }
    const float inv = 1.0f / fmaxf(sqrtf(red[0]), 1e-12f);

    __hip_bfloat16 o[4];
    o[0] = __float2bfloat16(v.x * inv);
    o[1] = __float2bfloat16(v.y * inv);
    o[2] = __float2bfloat16(v.z * inv);
    o[3] = __float2bfloat16(v.w * inv);
    uint2 packed;
    __builtin_memcpy(&packed, o, 8);
    *(uint2*)(dst + (size_t)row * D_SZ + t * 4) = packed;
}

// ---------------- Kernel 2: diagonal terms ----------------
__global__ __launch_bounds__(256) void diag_kernel(const unsigned short* __restrict__ zi,
                                                   const unsigned short* __restrict__ zj,
                                                   float* __restrict__ pos,
                                                   float* __restrict__ negd) {
    const int row = blockIdx.x;
    const int t = threadIdx.x;
    const __hip_bfloat16* a = (const __hip_bfloat16*)(zi + (size_t)row * D_SZ) + t * 4;
    const __hip_bfloat16* b = (const __hip_bfloat16*)(zj + (size_t)row * D_SZ) + t * 4;
    float s = 0.0f;
#pragma unroll
    for (int i = 0; i < 4; ++i)
        s += __bfloat162float(a[i]) * __bfloat162float(b[i]);

    __shared__ float red[256];
    red[t] = s;
    __syncthreads();
#pragma unroll
    for (int st = 128; st > 0; st >>= 1) {
        if (t < st) red[t] += red[t + st];
        __syncthreads();
    }
    if (t == 0) {
        float d = red[0];
        pos[row]  = softplus_f(TEMP - d);
        negd[row] = softplus_f(d - TEMP);
    }
}

// ------- Kernel 3: TDM-staged, double-buffered WMMA GEMM + softplus + reduce -------
__global__ __launch_bounds__(256) void gemm_softplus_kernel(const unsigned short* __restrict__ zj, // A rows (M)
                                                            const unsigned short* __restrict__ zi, // B rows (N)
                                                            float* __restrict__ partial) {
    // [buffer][operand(A=0,B=1)][128 rows * PITCH]
    __shared__ __align__(16) unsigned short tiles[2][2][128 * PITCH];
    __shared__ float red[256];

    const int t    = threadIdx.x;
    const int lane = t & 31;
    const int wave = t >> 5;        // 0..7
    const int wm   = wave & 3;      // 4 M groups of 32 rows
    const int wn   = wave >> 2;     // 2 N groups of 64 cols
    const int r    = lane & 15;
    const int h    = lane >> 4;

    const int mBase = blockIdx.y * 128;
    const int nBase = blockIdx.x * 128;

    v8f acc[2][4];
#pragma unroll
    for (int i = 0; i < 2; ++i)
#pragma unroll
        for (int j = 0; j < 4; ++j) { v8f z = {}; acc[i][j] = z; }

    // Prologue: DMA first K-stage into buffer 0 (wave 0 only; TDM ignores EXEC,
    // issuing from >1 wave would duplicate the transfer).
    if (wave == 0) {
        tdm_load_tile(zj + (size_t)mBase * D_SZ, (unsigned)(uintptr_t)&tiles[0][0][0]);
        tdm_load_tile(zi + (size_t)nBase * D_SZ, (unsigned)(uintptr_t)&tiles[0][1][0]);
        __builtin_amdgcn_s_wait_tensorcnt(0);
    }
    __syncthreads();

    for (int k0 = 0; k0 < D_SZ; k0 += KC) {
        const int buf = (k0 / KC) & 1;

        // Kick off DMA for next stage into the other buffer (overlaps WMMA below).
        if (k0 + KC < D_SZ && wave == 0) {
            tdm_load_tile(zj + (size_t)mBase * D_SZ + (k0 + KC),
                          (unsigned)(uintptr_t)&tiles[buf ^ 1][0][0]);
            tdm_load_tile(zi + (size_t)nBase * D_SZ + (k0 + KC),
                          (unsigned)(uintptr_t)&tiles[buf ^ 1][1][0]);
        }

        const unsigned short* Abuf = &tiles[buf][0][0];
        const unsigned short* Bbuf = &tiles[buf][1][0];

        FragAB a[2], b[4];
        // A frag (16x32 MxK): lane half h -> K=8h..8h+7 and K=16+8h..23+8h, contiguous b128s
#pragma unroll
        for (int i = 0; i < 2; ++i) {
            const int base = (wm * 32 + i * 16 + r) * PITCH + 8 * h;
            a[i].q[0] = *(const uint4*)&Abuf[base];
            a[i].q[1] = *(const uint4*)&Abuf[base + 16];
        }
        // B frag (32x16 KxN): lane half h -> K=8h..8h+7 and K=16+8h..23+8h for its column
#pragma unroll
        for (int j = 0; j < 4; ++j) {
            const int base = (wn * 64 + j * 16 + r) * PITCH + 8 * h;
            b[j].q[0] = *(const uint4*)&Bbuf[base];
            b[j].q[1] = *(const uint4*)&Bbuf[base + 16];
        }
#pragma unroll
        for (int i = 0; i < 2; ++i)
#pragma unroll
            for (int j = 0; j < 4; ++j)
                acc[i][j] = __builtin_amdgcn_wmma_f32_16x16x32_bf16(
                    false, a[i].v, false, b[j].v, (short)0, acc[i][j], false, false);

        __syncthreads();                       // all waves done reading tiles[buf]
        if (k0 + KC < D_SZ) {
            if (wave == 0) __builtin_amdgcn_s_wait_tensorcnt(0); // next buffer landed
            __syncthreads();
        }
    }

    // Fused epilogue: sum log1p(exp(sim - T)) over this 128x128 tile
    float s = 0.0f;
#pragma unroll
    for (int i = 0; i < 2; ++i)
#pragma unroll
        for (int j = 0; j < 4; ++j)
#pragma unroll
            for (int e = 0; e < 8; ++e)
                s += softplus_f(acc[i][j][e] - TEMP);

    red[t] = s;
    __syncthreads();
#pragma unroll
    for (int st = 128; st > 0; st >>= 1) {
        if (t < st) red[t] += red[t + st];
        __syncthreads();
    }
    if (t == 0) partial[blockIdx.y * 32 + blockIdx.x] = red[0];
}

// ---------------- Kernel 4: deterministic final reduction ----------------
__global__ __launch_bounds__(256) void final_kernel(const float* __restrict__ partial,
                                                    const float* __restrict__ pos,
                                                    const float* __restrict__ negd,
                                                    float* __restrict__ out) {
    const int t = threadIdx.x;
    float sAll = 0.0f, sPos = 0.0f, sNegD = 0.0f;
    for (int i = t; i < 1024; i += 256) sAll += partial[i];
    for (int i = t; i < B_SZ; i += 256) { sPos += pos[i]; sNegD += negd[i]; }

    __shared__ float r0[256], r1[256], r2[256];
    r0[t] = sAll; r1[t] = sPos; r2[t] = sNegD;
    __syncthreads();
#pragma unroll
    for (int st = 128; st > 0; st >>= 1) {
        if (t < st) { r0[t] += r0[t + st]; r1[t] += r1[t + st]; r2[t] += r2[t + st]; }
        __syncthreads();
    }
    if (t == 0) {
        const float Bf = (float)B_SZ;
        float positives_sum = r1[0] / Bf;
        float negatives_sum = (r0[0] - r2[0]) / (Bf * (Bf - 1.0f));
        out[0] = HGAMMA * positives_sum + (1.0f - HGAMMA) * negatives_sum;
    }
}

extern "C" void kernel_launch(void* const* d_in, const int* in_sizes, int n_in,
                              void* d_out, int out_size, void* d_ws, size_t ws_size,
                              hipStream_t stream) {
    const float* emb_i = (const float*)d_in[0];
    const float* emb_j = (const float*)d_in[1];
    float* out = (float*)d_out;

    char* ws = (char*)d_ws;
    unsigned short* zi = (unsigned short*)ws;                                  // 8 MB
    unsigned short* zj = (unsigned short*)(ws + (size_t)B_SZ * D_SZ * 2);      // 8 MB
    float* pos     = (float*)(ws + (size_t)B_SZ * D_SZ * 4);                   // 16 KB
    float* negd    = pos + B_SZ;                                               // 16 KB
    float* partial = negd + B_SZ;                                              // 4 KB

    norm_kernel<<<B_SZ, 256, 0, stream>>>(emb_i, zi);
    norm_kernel<<<B_SZ, 256, 0, stream>>>(emb_j, zj);
    diag_kernel<<<B_SZ, 256, 0, stream>>>(zi, zj, pos, negd);

    dim3 grid(32, 32);  // N tiles x, M tiles y (4096/128)
    gemm_softplus_kernel<<<grid, 256, 0, stream>>>(zj, zi, partial);

    final_kernel<<<1, 256, 0, stream>>>(partial, pos, negd, out);
}
